// GNNStack_40441412059412
// MI455X (gfx1250) — matrix operand order repaired and checked by reading, not compiled, not run on hardware
//
#include <hip/hip_runtime.h>
#include <hip/hip_bf16.h>
#include <math.h>

#define N_NODES 100000
#define N_EDGES 1600000
#define DIM     128
#define NGRAPH  512
#define NCLASS  10

typedef __attribute__((ext_vector_type(16))) __bf16 v16bf;
typedef __attribute__((ext_vector_type(8)))  float  v8f;

union BF16x16 { uint4 q[2]; unsigned short u[16]; v16bf v; };

__device__ __forceinline__ unsigned short f2bf(float f) {
  return __builtin_bit_cast(unsigned short, (__bf16)f);  // native f32->bf16 (RNE)
}

// A-fragment (16x32 bf16, MxK) from a wave-private 16x128 bf16 LDS tile.
// Lanes 0-15: M=lane, K=K0+0..15; lanes 16-31: M=lane-16, K=K0+16..31.
// Both 8-element runs are 16B-contiguous & 16B-aligned -> two ds_load_b128.
__device__ __forceinline__ v16bf load_afrag(const unsigned short* tile, int lane, int K0) {
  const int m  = lane & 15;
  const int kb = K0 + ((lane >> 4) << 4);
  const unsigned short* row = tile + m * DIM + kb;
  BF16x16 f;
  f.q[0] = *(const uint4*)(row);        // K = kb .. kb+7
  f.q[1] = *(const uint4*)(row + 8);    // K = kb+8 .. kb+15
  return f.v;
}

// B-fragment (32x16 bf16, KxN) from LDS weights pre-swizzled into fragment order:
// block (n,kc) -> 32 lanes x 16 contiguous bf16 values -> two ds_load_b128 per lane.
__device__ __forceinline__ v16bf load_bfrag_sw(const unsigned short* sWsw, int lane, int kc, int n) {
  const unsigned short* p = sWsw + ((((n << 2) + kc) << 5) + lane) * 16;
  BF16x16 f;
  f.q[0] = *(const uint4*)(p);
  f.q[1] = *(const uint4*)(p + 8);
  return f.v;
}

// Stage a 128x128 f32 weight matrix from global into LDS via async copies
// (GLOBAL_LOAD_ASYNC_TO_LDS_B128, ASYNCcnt), then convert/swizzle to bf16
// fragment layout. Two 32 KB halves through sStage.
__device__ __forceinline__ void load_weight_swizzled(
    const float* __restrict__ w, unsigned short* sWsw, float* sStage, int tid)
{
  const unsigned stageBase = (unsigned)(size_t)sStage;  // low 32 bits = LDS offset
  for (int h = 0; h < 2; ++h) {
    const float* gsrc = w + h * 8192;
#pragma unroll
    for (int j = 0; j < 8; ++j) {
      const unsigned lds = stageBase + (unsigned)((j * 256 + tid) * 16);
      const float* gp = gsrc + (j * 256 + tid) * 4;
      asm volatile("global_load_async_to_lds_b128 %0, %1, off"
                   :: "v"(lds), "v"(gp) : "memory");
    }
    asm volatile("s_wait_asynccnt 0x0" ::: "memory");
    __syncthreads();
    const int k0 = h * 64;
    for (int i = tid; i < 8192; i += 256) {
      const int k = k0 + (i >> 7), col = i & 127;
      const int kc = k >> 5, krem = k & 31;
      const int lane = ((krem >> 4) << 4) | (col & 15);
      const int idx = (((((col >> 4) << 2) + kc) << 5) + lane) * 16 + (krem & 15);
      sWsw[idx] = f2bf(sStage[i]);
    }
    __syncthreads();
  }
}

// Fused GIN MLP: out = relu(z @ w1 + b1) @ w2 + b2  (hidden never leaves LDS).
// outRelu2 (optional) mirrors outRelu: pre-initializes next layer's z buffer
// (safe in-place vs. input z: each block writes only rows it already read).
__global__ __launch_bounds__(256) void gin_mlp_kernel(
    const float* __restrict__ z,
    const float* __restrict__ w1, const float* __restrict__ b1,
    const float* __restrict__ w2, const float* __restrict__ b2,
    float* __restrict__ outRelu, float* __restrict__ outRelu2,
    float* __restrict__ outRaw, int nrows)
{
  __shared__ unsigned short sWsw[DIM * DIM];      // 32 KB swizzled bf16 weights (w1 then w2)
  __shared__ unsigned short sTile[8 * 16 * DIM];  // 32 KB per-wave 16x128 bf16 tiles
  __shared__ float sStage[8192];                  // 32 KB f32 async staging buffer

  const int tid  = threadIdx.x;
  const int wave = tid >> 5;
  const int lane = tid & 31;
  unsigned short* tile = sTile + wave * 16 * DIM;

  // Phase 1: w1 -> LDS (async stage + bf16 swizzle)
  load_weight_swizzled(w1, sWsw, sStage, tid);

  const int rbase = blockIdx.x * 128 + wave * 16;

  // Wave-private input tile: 16 rows x 128 cols, f32 global -> bf16 LDS (ds_store_b64)
  for (int r = 0; r < 16; ++r) {
    const int row = rbase + r;
    float4 v = make_float4(0.f, 0.f, 0.f, 0.f);
    if (row < nrows) v = *(const float4*)(z + (size_t)row * DIM + lane * 4);
    ushort4 s;
    s.x = f2bf(v.x); s.y = f2bf(v.y); s.z = f2bf(v.z); s.w = f2bf(v.w);
    *(ushort4*)(tile + r * DIM + lane * 4) = s;
  }

  v16bf a0 = load_afrag(tile, lane, 0);
  v16bf a1 = load_afrag(tile, lane, 32);
  v16bf a2 = load_afrag(tile, lane, 64);
  v16bf a3 = load_afrag(tile, lane, 96);

  const int colb   = lane & 15;
  const int rowoff = (lane >> 4) << 3;

  // GEMM1 + bias + ReLU -> hidden tile back into wave-private LDS (bf16)
#pragma unroll
  for (int n = 0; n < 8; ++n) {
    const int col = (n << 4) + colb;
    const float bv = b1[col];
    v8f acc;
#pragma unroll
    for (int i = 0; i < 8; ++i) acc[i] = bv;
    acc = __builtin_amdgcn_wmma_f32_16x16x32_bf16(false, a0, false, load_bfrag_sw(sWsw, lane, 0, n), (short)0, acc, false, false);
    acc = __builtin_amdgcn_wmma_f32_16x16x32_bf16(false, a1, false, load_bfrag_sw(sWsw, lane, 1, n), (short)0, acc, false, false);
    acc = __builtin_amdgcn_wmma_f32_16x16x32_bf16(false, a2, false, load_bfrag_sw(sWsw, lane, 2, n), (short)0, acc, false, false);
    acc = __builtin_amdgcn_wmma_f32_16x16x32_bf16(false, a3, false, load_bfrag_sw(sWsw, lane, 3, n), (short)0, acc, false, false);
#pragma unroll
    for (int i = 0; i < 8; ++i)
      tile[(rowoff + i) * DIM + col] = f2bf(fmaxf(acc[i], 0.f));  // D layout: row=i+rowoff
  }

  __syncthreads();  // all waves done reading w1 fragments
  // Phase 2: w2 -> LDS (same buffers)
  load_weight_swizzled(w2, sWsw, sStage, tid);

  a0 = load_afrag(tile, lane, 0);
  a1 = load_afrag(tile, lane, 32);
  a2 = load_afrag(tile, lane, 64);
  a3 = load_afrag(tile, lane, 96);

#pragma unroll
  for (int n = 0; n < 8; ++n) {
    const int col = (n << 4) + colb;
    const float bv = b2[col];
    v8f acc;
#pragma unroll
    for (int i = 0; i < 8; ++i) acc[i] = bv;
    acc = __builtin_amdgcn_wmma_f32_16x16x32_bf16(false, a0, false, load_bfrag_sw(sWsw, lane, 0, n), (short)0, acc, false, false);
    acc = __builtin_amdgcn_wmma_f32_16x16x32_bf16(false, a1, false, load_bfrag_sw(sWsw, lane, 1, n), (short)0, acc, false, false);
    acc = __builtin_amdgcn_wmma_f32_16x16x32_bf16(false, a2, false, load_bfrag_sw(sWsw, lane, 2, n), (short)0, acc, false, false);
    acc = __builtin_amdgcn_wmma_f32_16x16x32_bf16(false, a3, false, load_bfrag_sw(sWsw, lane, 3, n), (short)0, acc, false, false);
#pragma unroll
    for (int i = 0; i < 8; ++i) {
      const int row = rbase + rowoff + i;
      if (row < nrows) {
        const float v = acc[i];
        const float rv = fmaxf(v, 0.f);
        const size_t o = (size_t)row * DIM + col;
        if (outRaw)   outRaw[o]   = v;
        outRelu[o] = rv;
        if (outRelu2) outRelu2[o] = rv;
      }
    }
  }
}

// One wave per edge: z[dst] += h[src]   (z pre-initialized with h -> GIN self term)
__global__ __launch_bounds__(256) void scatter_add_kernel(
    const float* __restrict__ h, const int* __restrict__ src, const int* __restrict__ dst,
    float* __restrict__ z, int nedges)
{
  const int e = blockIdx.x * 8 + (threadIdx.x >> 5);
  const int lane = threadIdx.x & 31;
  if (e >= nedges) return;
  // lookahead prefetch of a future source row (global_prefetch_b8)
  const int e2 = e + 512;
  if (e2 < nedges) {
    const int s2 = src[e2];
    __builtin_prefetch(h + (size_t)s2 * DIM + lane * 4, 0, 1);
  }
  const int s = src[e], d = dst[e];
  const float4 v = *(const float4*)(h + (size_t)s * DIM + lane * 4);
  float* zp = z + (size_t)d * DIM + lane * 4;
  atomicAdd(zp + 0, v.x); atomicAdd(zp + 1, v.y);
  atomicAdd(zp + 2, v.z); atomicAdd(zp + 3, v.w);
}

__global__ void zero_kernel(float* __restrict__ p, int n) {
  const int i = blockIdx.x * blockDim.x + threadIdx.x;
  if (i < n) p[i] = 0.f;
}

// One wave per node: pooled[batch[n]] += h[n]; cnt[batch[n]] += 1
__global__ __launch_bounds__(256) void pool_acc_kernel(
    const float* __restrict__ h, const int* __restrict__ batch,
    float* __restrict__ pooled, float* __restrict__ cnt, int nrows)
{
  const int node = blockIdx.x * 8 + (threadIdx.x >> 5);
  const int lane = threadIdx.x & 31;
  if (node >= nrows) return;
  const int g = batch[node];
  const float4 v = *(const float4*)(h + (size_t)node * DIM + lane * 4);
  float* pp = pooled + (size_t)g * DIM + lane * 4;
  atomicAdd(pp + 0, v.x); atomicAdd(pp + 1, v.y);
  atomicAdd(pp + 2, v.z); atomicAdd(pp + 3, v.w);
  if (lane == 0) atomicAdd(cnt + g, 1.0f);
}

__global__ void pool_div_kernel(float* __restrict__ pooled, const float* __restrict__ cnt) {
  const int i = blockIdx.x * blockDim.x + threadIdx.x;
  if (i < NGRAPH * DIM) pooled[i] /= fmaxf(cnt[i >> 7], 1.0f);
}

// hidden = pooled @ wp1 + bp1   (512x128 @ 128x128)
__global__ void post1_kernel(const float* __restrict__ pooled, const float* __restrict__ wp1,
                             const float* __restrict__ bp1, float* __restrict__ hidden) {
  const int i = blockIdx.x * blockDim.x + threadIdx.x;
  if (i >= NGRAPH * DIM) return;
  const int g = i >> 7, c = i & 127;
  const float* pr = pooled + (size_t)g * DIM;
  float acc = bp1[c];
  for (int k = 0; k < DIM; ++k) acc = fmaf(pr[k], wp1[k * DIM + c], acc);
  hidden[i] = acc;
}

// logits = hidden @ wp2 + bp2, then log_softmax per graph (one wave32 per graph)
__global__ __launch_bounds__(32) void post2_kernel(
    const float* __restrict__ hidden, const float* __restrict__ wp2,
    const float* __restrict__ bp2, float* __restrict__ out)
{
  __shared__ float srow[DIM];
  __shared__ float sl[NCLASS];
  __shared__ float sred[2];
  const int g = blockIdx.x, t = threadIdx.x;
  for (int i = t; i < DIM; i += 32) srow[i] = hidden[(size_t)g * DIM + i];
  __syncthreads();
  if (t < NCLASS) {
    float acc = bp2[t];
    for (int k = 0; k < DIM; ++k) acc = fmaf(srow[k], wp2[k * NCLASS + t], acc);
    sl[t] = acc;
  }
  __syncthreads();
  if (t == 0) {
    float m = sl[0];
    for (int i = 1; i < NCLASS; ++i) m = fmaxf(m, sl[i]);
    float s = 0.f;
    for (int i = 0; i < NCLASS; ++i) s += expf(sl[i] - m);
    sred[0] = m; sred[1] = logf(s);
  }
  __syncthreads();
  if (t < NCLASS) out[(size_t)g * NCLASS + t] = sl[t] - sred[0] - sred[1];
}

extern "C" void kernel_launch(void* const* d_in, const int* in_sizes, int n_in,
                              void* d_out, int out_size, void* d_ws, size_t ws_size,
                              hipStream_t stream)
{
  (void)in_sizes; (void)n_in; (void)out_size; (void)ws_size;

  const float* x     = (const float*)d_in[0];
  const int*   ei    = (const int*)d_in[1];
  const int*   batch = (const int*)d_in[2];
  const float* w1[3] = {(const float*)d_in[3],  (const float*)d_in[7],  (const float*)d_in[11]};
  const float* b1[3] = {(const float*)d_in[4],  (const float*)d_in[8],  (const float*)d_in[12]};
  const float* w2[3] = {(const float*)d_in[5],  (const float*)d_in[9],  (const float*)d_in[13]};
  const float* b2[3] = {(const float*)d_in[6],  (const float*)d_in[10], (const float*)d_in[14]};
  const float* wp1 = (const float*)d_in[15];
  const float* bp1 = (const float*)d_in[16];
  const float* wp2 = (const float*)d_in[17];
  const float* bp2 = (const float*)d_in[18];

  const size_t NH = (size_t)N_NODES * DIM;
  float* bufA   = (float*)d_ws;                 // node features (post-ReLU)
  float* bufB   = bufA + NH;                    // z = h + scatter(h)
  float* pooled = bufB + NH;                    // G x 128
  float* cnt    = pooled + (size_t)NGRAPH * DIM;
  float* hidden = cnt + NGRAPH;                 // G x 128

  float* embOut = (float*)d_out;
  float* lsmOut = embOut + NH;

  const int mlpBlocks  = (N_NODES + 127) / 128;
  const int edgeBlocks = (N_EDGES + 7) / 8;
  const int nodeBlocks = (N_NODES + 7) / 8;
  const int* src = ei;
  const int* dst = ei + N_EDGES;

  // Layer 0: z-init via d2d copy of x; layers 1/2 get z pre-initialized by the
  // previous MLP kernel's dual ReLU write (saves two 51 MB passes).
  hipMemcpyAsync(bufB, x, NH * sizeof(float), hipMemcpyDeviceToDevice, stream);
  scatter_add_kernel<<<edgeBlocks, 256, 0, stream>>>(x, src, dst, bufB, N_EDGES);
  gin_mlp_kernel<<<mlpBlocks, 256, 0, stream>>>(bufB, w1[0], b1[0], w2[0], b2[0],
                                                bufA, bufB, nullptr, N_NODES);

  scatter_add_kernel<<<edgeBlocks, 256, 0, stream>>>(bufA, src, dst, bufB, N_EDGES);
  gin_mlp_kernel<<<mlpBlocks, 256, 0, stream>>>(bufB, w1[1], b1[1], w2[1], b2[1],
                                                bufA, bufB, nullptr, N_NODES);

  scatter_add_kernel<<<edgeBlocks, 256, 0, stream>>>(bufA, src, dst, bufB, N_EDGES);
  gin_mlp_kernel<<<mlpBlocks, 256, 0, stream>>>(bufB, w1[2], b1[2], w2[2], b2[2],
                                                bufA, nullptr, embOut, N_NODES);

  zero_kernel<<<(NGRAPH * DIM + NGRAPH + 255) / 256, 256, 0, stream>>>(pooled, NGRAPH * DIM + NGRAPH);
  pool_acc_kernel<<<nodeBlocks, 256, 0, stream>>>(bufA, batch, pooled, cnt, N_NODES);
  pool_div_kernel<<<(NGRAPH * DIM + 255) / 256, 256, 0, stream>>>(pooled, cnt);
  post1_kernel<<<(NGRAPH * DIM + 255) / 256, 256, 0, stream>>>(pooled, wp1, bp1, hidden);
  post2_kernel<<<NGRAPH, 32, 0, stream>>>(hidden, wp2, bp2, lsmOut);
}
